// StateBank_37847251812684
// MI455X (gfx1250) — matrix operand order repaired and checked by reading, not compiled, not run on hardware
//
#include <hip/hip_runtime.h>

typedef unsigned short ushort_t;
typedef __attribute__((ext_vector_type(16))) __bf16 v16bf;
typedef __attribute__((ext_vector_type(8)))  float  v8f;

// ---------------- problem constants ----------------
#define BB 4
#define TT 2048
#define DD 1024
#define KK 4
#define EE (KK * DD)             // 4096
#define MM (BB * TT)             // 8192
#define GSIZE ((size_t)MM * DD)  // g output elements: 8,388,608

// ---------------- GEMM tiling ----------------
#define KS   32                   // bf16 k-elements per pipeline stage
#define ROWB 80                   // LDS bytes per tile row: 64 data + 16 pad (16B-aligned, conflict-free)
#define TILEBYTES (128 * ROWB)    // 10,240 B per 128x32 tile

// ---------------- helpers ----------------
__device__ __forceinline__ ushort_t f2bf(float f) {
  unsigned int u = __float_as_uint(f);
  u += 0x7FFFu + ((u >> 16) & 1u);   // round-to-nearest-even
  return (ushort_t)(u >> 16);
}
__device__ __forceinline__ float bf2f(ushort_t h) {
  return __uint_as_float(((unsigned int)h) << 16);
}

union Frag {
  uint4  u[2];
  v16bf  v;
};

// Async global->LDS copy: per-lane 16B, one wave instruction moves 512B.
// VDST = LDS byte offset (low 32 bits of flat shared address), VADDR = 64-bit global addr.
__device__ __forceinline__ void async_b128(unsigned lds_off, const void* g) {
  asm volatile("global_load_async_to_lds_b128 %0, %1, off"
               :: "v"(lds_off), "v"((unsigned long long)(uintptr_t)g)
               : "memory");
}

// ---------------- f32 -> bf16 conversion (vectorized, grid-stride) ----------------
__global__ void cvt_f32_bf16_kernel(const float* __restrict__ src,
                                    ushort_t* __restrict__ dst, size_t n4) {
  size_t i = (size_t)blockIdx.x * blockDim.x + threadIdx.x;
  size_t stride = (size_t)gridDim.x * blockDim.x;
  for (; i < n4; i += stride) {
    float4 f = ((const float4*)src)[i];
    ushort4 o;
    o.x = f2bf(f.x); o.y = f2bf(f.y); o.z = f2bf(f.z); o.w = f2bf(f.w);
    ((ushort4*)dst)[i] = o;
  }
}

// ---------------- bf16 WMMA GEMM, async double-buffered LDS staging ----------------
// C[M,N] = A[M,Kd] * Bt[N,Kd]^T + bias[N]
// A, Bt row-major bf16, reduction dim contiguous.
// Block: 256 threads = 8 waves; block tile 128x128; wave tile 32x64 (2x4 16x16 subtiles).
template <bool OUT_BF16>
__global__ __launch_bounds__(256)
void gemm_bf16_wmma_kernel(const ushort_t* __restrict__ A,
                           const ushort_t* __restrict__ Bt,
                           const float* __restrict__ bias,
                           float* __restrict__ Cf,
                           ushort_t* __restrict__ Cb,
                           int M, int N, int Kd) {
  __shared__ char smem[2 * 2 * TILEBYTES];   // [buf][A|B][128 rows][ROWB] = 40 KB
  const unsigned smem_base = (unsigned)(uintptr_t)(void*)smem;  // LDS byte offset

  const int tid = threadIdx.x;
  const int blockRow = blockIdx.y * 128;
  const int blockCol = blockIdx.x * 128;

  // Stage issue: 128x32 A tile + 128x32 B tile = 2 * 512 chunks of 16B.
  // 256 lanes * 2 iterations -> each wave issues 4 async instructions per stage.
  auto issue_stage = [&](int kk, int buf) {
    const unsigned baseA = smem_base + (unsigned)buf * (2 * TILEBYTES);
    const unsigned baseB = baseA + TILEBYTES;
#pragma unroll
    for (int i = 0; i < 2; ++i) {
      const int chunk = tid + i * 256;   // 0..511
      const int row = chunk >> 2;        // 0..127
      const int cp  = chunk & 3;         // 16B chunk within 64B row
      async_b128(baseA + (unsigned)(row * ROWB + cp * 16),
                 A + (size_t)(blockRow + row) * Kd + kk + cp * 8);
      async_b128(baseB + (unsigned)(row * ROWB + cp * 16),
                 Bt + (size_t)(blockCol + row) * Kd + kk + cp * 8);
    }
  };

  const int lane = tid & 31;
  const int wave = tid >> 5;
  const int l    = lane & 15;   // row/col within a 16x16 tile
  const int half = lane >> 4;   // which K-half this lane holds
  const int wr   = (wave & 3) * 32;   // wave row base within block tile
  const int wc   = (wave >> 2) * 64;  // wave col base within block tile

  v8f acc[2][4];
#pragma unroll
  for (int i = 0; i < 2; ++i)
#pragma unroll
    for (int j = 0; j < 4; ++j)
      acc[i][j] = (v8f)(0.0f);

  const int nsteps = Kd / KS;
  issue_stage(0, 0);

  for (int s = 0; s < nsteps; ++s) {
    if (s + 1 < nsteps) {
      issue_stage((s + 1) * KS, (s + 1) & 1);
      asm volatile("s_wait_asynccnt 0x4" ::: "memory");  // stage s complete, s+1 in flight
    } else {
      asm volatile("s_wait_asynccnt 0x0" ::: "memory");
    }
    __syncthreads();   // tile (all waves' copies) visible

    const char* bufA = smem + (s & 1) * (2 * TILEBYTES);
    const char* bufB = bufA + TILEBYTES;

    Frag a[2], b[4];
    // A fragment 16x32: lane l holds row; elems 0..7 -> K=half*8.., elems 8..15 -> K=16+half*8..
#pragma unroll
    for (int i = 0; i < 2; ++i) {
      const char* p = bufA + (wr + 16 * i + l) * ROWB + half * 16;
      a[i].u[0] = *(const uint4*)(p);
      a[i].u[1] = *(const uint4*)(p + 32);
    }
    // B fragment 32x16: lane l holds column; elems 0..7 -> K=half*8.., elems 8..15 -> K=16+half*8..
#pragma unroll
    for (int j = 0; j < 4; ++j) {
      const char* p = bufB + (wc + 16 * j + l) * ROWB + half * 16;
      b[j].u[0] = *(const uint4*)(p);
      b[j].u[1] = *(const uint4*)(p + 32);
    }
#pragma unroll
    for (int i = 0; i < 2; ++i)
#pragma unroll
      for (int j = 0; j < 4; ++j)
        acc[i][j] = __builtin_amdgcn_wmma_f32_16x16x32_bf16(
            false, a[i].v, false, b[j].v, (short)0, acc[i][j], false, false);

    __syncthreads();   // safe to overwrite buf (s&1) by stage s+2's async writes
  }

  // Epilogue: C/D layout -> VGPR r: M = r + (half?8:0), N = l within each 16x16 tile
#pragma unroll
  for (int i = 0; i < 2; ++i) {
    const int row0 = blockRow + wr + 16 * i + half * 8;
#pragma unroll
    for (int j = 0; j < 4; ++j) {
      const int col = blockCol + wc + 16 * j + l;
      const float bv = bias ? bias[col] : 0.0f;
#pragma unroll
      for (int r = 0; r < 8; ++r) {
        const float v = acc[i][j][r] + bv;
        const size_t idx = (size_t)(row0 + r) * N + col;
        if (OUT_BF16) Cb[idx] = f2bf(v);
        else          Cf[idx] = v;
      }
    }
  }
}

// ---------------- decaying-state scan over T ----------------
// x: (B,T,E) bf16 ; writes s_seq (B,T,E) bf16 and s_last (B,E) f32.
// One thread per channel (b,e); lanes contiguous in e -> coalesced at every t.
__global__ __launch_bounds__(256)
void scan_kernel(const ushort_t* __restrict__ x,
                 const float* __restrict__ s0,
                 const float* __restrict__ decay_logit,
                 ushort_t* __restrict__ s_seq,
                 float* __restrict__ s_last) {
  const int ch = blockIdx.x * blockDim.x + threadIdx.x;  // 0 .. B*E-1
  const int b = ch / EE;
  const int e = ch - b * EE;
  const int k = e / DD;

  const float dec = 1.0f / (1.0f + __expf(-decay_logit[k]));
  float s = s0[(size_t)b * EE + e];

  size_t idx = (size_t)b * TT * EE + e;
#pragma unroll 4
  for (int t = 0; t < TT; ++t, idx += EE) {
    s = __fmaf_rn(dec, s, bf2f(x[idx]));
    s_seq[idx] = f2bf(s);
  }
  s_last[(size_t)b * EE + e] = s;
}

// ---------------- launcher ----------------
extern "C" void kernel_launch(void* const* d_in, const int* in_sizes, int n_in,
                              void* d_out, int out_size, void* d_ws, size_t ws_size,
                              hipStream_t stream) {
  const float* u           = (const float*)d_in[0];  // (B,T,D)
  const float* s0          = (const float*)d_in[1];  // (B,K,D)
  const float* W_in        = (const float*)d_in[2];  // (K*D, D)
  const float* b_in        = (const float*)d_in[3];  // (K*D,)
  const float* W_out       = (const float*)d_in[4];  // (D, K*D)
  const float* b_out       = (const float*)d_in[5];  // (D,)
  const float* decay_logit = (const float*)d_in[6];  // (K,)

  float* out_g      = (float*)d_out;          // (B,T,D)
  float* out_s_last = (float*)d_out + GSIZE;  // (B,K,D)

  // workspace layout (bytes), all 256-aligned; total 160 MB
  char* ws = (char*)d_ws;
  ushort_t* u_bf    = (ushort_t*)(ws);                  // 16 MB : 8192x1024
  ushort_t* win_bf  = (ushort_t*)(ws + (16u << 20));    //  8 MB : 4096x1024
  ushort_t* wout_bf = (ushort_t*)(ws + (24u << 20));    //  8 MB : 1024x4096
  ushort_t* inp_bf  = (ushort_t*)(ws + (32u << 20));    // 64 MB : 8192x4096
  ushort_t* sseq_bf = (ushort_t*)(ws + (96u << 20));    // 64 MB : 8192x4096

  // 1) convert operands to bf16
  {
    const size_t n_u = (size_t)MM * DD / 4;  // float4 groups
    const size_t n_w = (size_t)EE * DD / 4;
    cvt_f32_bf16_kernel<<<1024, 256, 0, stream>>>(u, u_bf, n_u);
    cvt_f32_bf16_kernel<<<1024, 256, 0, stream>>>(W_in, win_bf, n_w);
    cvt_f32_bf16_kernel<<<1024, 256, 0, stream>>>(W_out, wout_bf, n_w);
  }

  // 2) GEMM1: inp(8192x4096) = u(8192x1024) x W_in^T + b_in  -> bf16
  {
    dim3 grid(EE / 128, MM / 128);  // (32, 64)
    gemm_bf16_wmma_kernel<true><<<grid, 256, 0, stream>>>(
        u_bf, win_bf, b_in, nullptr, inp_bf, MM, EE, DD);
  }

  // 3) scan over T (f32 state, bf16 in/out), also emits s_last (f32)
  {
    const int nch = BB * EE;  // 16384
    scan_kernel<<<nch / 256, 256, 0, stream>>>(inp_bf, s0, decay_logit,
                                               sseq_bf, out_s_last);
  }

  // 4) GEMM2: g(8192x1024) = s_seq(8192x4096) x W_out^T + b_out -> f32 to d_out
  {
    dim3 grid(DD / 128, MM / 128);  // (8, 64)
    gemm_bf16_wmma_kernel<false><<<grid, 256, 0, stream>>>(
        sseq_bf, wout_bf, b_out, out_g, nullptr, MM, DD, EE);
  }
}
